// Attention_43353399886449
// MI455X (gfx1250) — matrix operand order repaired and checked by reading, not compiled, not run on hardware
//
#include <hip/hip_runtime.h>

// Problem constants (from the reference)
#define B_  2
#define S_  2048
#define D_  2048
#define H_  16
#define KV_ 4
#define HD_ 128
#define TOK_ (B_ * S_)   // 4096

typedef __attribute__((ext_vector_type(16))) _Float16 v16h;
typedef __attribute__((ext_vector_type(8)))  float    v8f;

static __device__ __forceinline__ v8f wmma16(v16h a, v16h b, v8f c) {
  // D = A(16x32 f16) * B(32x16 f16) + C(f32)
  return __builtin_amdgcn_wmma_f32_16x16x32_f16(false, a, false, b, 0, c, false, false);
}

// A-fragment K index for element pair j (j=0..7), pair base (even):
//   k = (j>=4 ? 16 : 0) + half*8 + (j&3)*2      (16B-contiguous per quad)
#define KA(j, half) ((((j) & 4) << 2) + ((half) << 3) + (((j) & 3) << 1))
// B-fragment K index: same as A: k = (j>=4 ? 16 : 0) + half*8 + (j&3)*2
#define KB(j, half) ((((j) & 4) << 2) + ((half) << 3) + (((j) & 3) << 1))

// ---------------------------------------------------------------------------
// Tiled WMMA GEMM: C[MxN] = A[MxK] @ B[KxN]
// A: f32 or f16 row-major (ld = K); B: f32 row-major (ld = N); C: f16 or f32.
// Block tile 128x128, BK=32, double-buffered LDS.
// 256 threads = 8 waves (2x4); wave tile 64x32 -> 8 WMMAs / 6 frag loads.
// ---------------------------------------------------------------------------
template <bool A_HALF, bool OUT_HALF>
__global__ __launch_bounds__(256)
void gemm_wmma_kernel(const void* __restrict__ Ap, const float* __restrict__ Bp,
                      void* __restrict__ Cp, int M, int N, int K) {
  constexpr int BM = 128, BN = 128, BK = 32, LDT = 40;
  __shared__ _Float16 lsA[2][BM * LDT];   // [m][k]
  __shared__ _Float16 lsB[2][BN * LDT];   // transposed: [n][k]

  const int tid  = threadIdx.x;
  const int lane = tid & 31;
  const int wave = tid >> 5;
  const int wr = wave >> 2, wc = wave & 3;       // 2x4 wave grid
  const int m0 = blockIdx.y * BM, n0 = blockIdx.x * BN;
  const int nL = lane & 15, h2 = lane >> 4;

  v8f acc[4][2];
#pragma unroll
  for (int i = 0; i < 4; ++i)
#pragma unroll
    for (int j = 0; j < 2; ++j) acc[i][j] = (v8f){0, 0, 0, 0, 0, 0, 0, 0};

  // ---- tile stager: global (f32 or f16) -> f16 LDS buffers ----
  auto stage = [&](int k0, int buf) {
    // A tile: 128x32 -> lsA[buf][m][k]
#pragma unroll
    for (int i = 0; i < 4; ++i) {
      const int e = tid + i * 256;           // 1024 chunks of 4 elems
      const int row = e >> 3, seg = e & 7;
      union { _Float16 h[4]; unsigned long long u; } t;
      if (A_HALF) {
        t.u = *(const unsigned long long*)((const _Float16*)Ap +
              (size_t)(m0 + row) * K + k0 + seg * 4);
      } else {
        const float4 f = *(const float4*)((const float*)Ap +
              (size_t)(m0 + row) * K + k0 + seg * 4);
        t.h[0] = (_Float16)f.x; t.h[1] = (_Float16)f.y;
        t.h[2] = (_Float16)f.z; t.h[3] = (_Float16)f.w;
      }
      *(unsigned long long*)&lsA[buf][row * LDT + seg * 4] = t.u;
    }
    // B tile: 32x128 f32, transposed -> lsB[buf][n][k]
#pragma unroll
    for (int i = 0; i < 16; ++i) {
      const int e = tid + i * 256;           // 4096 elems
      const int kk = e >> 7, nn = e & 127;
      lsB[buf][nn * LDT + kk] = (_Float16)Bp[(size_t)(k0 + kk) * N + n0 + nn];
    }
  };

  stage(0, 0);
  __syncthreads();

  const int NK = K / BK;
  for (int kt = 0; kt < NK; ++kt) {
    const int cur = kt & 1;
    if (kt + 1 < NK) stage((kt + 1) * BK, cur ^ 1);   // prefetch next tile

    // ---- build fragments from current buffer and WMMA ----
    const _Float16* aB = &lsA[cur][(wr * 64) * LDT];
    const _Float16* bB = &lsB[cur][(wc * 32) * LDT];
    v16h af[4], bf[2];
#pragma unroll
    for (int t = 0; t < 4; ++t) {
      union { v16h v; unsigned u[8]; } r;
#pragma unroll
      for (int j = 0; j < 8; ++j)
        r.u[j] = *(const unsigned*)(aB + (t * 16 + nL) * LDT + KA(j, h2));
      af[t] = r.v;
    }
#pragma unroll
    for (int t = 0; t < 2; ++t) {
      union { v16h v; unsigned u[8]; } r;
#pragma unroll
      for (int j = 0; j < 8; ++j)
        r.u[j] = *(const unsigned*)(bB + (t * 16 + nL) * LDT + KB(j, h2));
      bf[t] = r.v;
    }
#pragma unroll
    for (int ti = 0; ti < 4; ++ti)
#pragma unroll
      for (int tj = 0; tj < 2; ++tj)
        acc[ti][tj] = wmma16(af[ti], bf[tj], acc[ti][tj]);

    __syncthreads();   // next-buffer stores complete; cur reads done
  }

  // ---- epilogue: C layout lane = n + 16*(m>=8), vgpr i = m%8 ----
#pragma unroll
  for (int ti = 0; ti < 4; ++ti)
#pragma unroll
    for (int tj = 0; tj < 2; ++tj)
#pragma unroll
      for (int i = 0; i < 8; ++i) {
        const int row = m0 + wr * 64 + ti * 16 + h2 * 8 + i;
        const int col = n0 + wc * 32 + tj * 16 + nL;
        if (OUT_HALF)
          ((_Float16*)Cp)[(size_t)row * N + col] = (_Float16)acc[ti][tj][i];
        else
          ((float*)Cp)[(size_t)row * N + col] = acc[ti][tj][i];
      }
}

// ---------------------------------------------------------------------------
// RoPE (in place on f16), output scale folds attention 1/sqrt(HD) into Q.
// X layout: [tok][heads*128]; one thread per (tok, head, d<64) pair.
// ---------------------------------------------------------------------------
__global__ __launch_bounds__(256)
void rope_kernel(_Float16* __restrict__ X, const float* __restrict__ cosT,
                 const float* __restrict__ sinT, int heads, float scale) {
  const size_t gid = (size_t)blockIdx.x * blockDim.x + threadIdx.x;
  const int d = (int)(gid & 63);
  const size_t th = gid >> 6;
  const int h = (int)(th % heads);
  const size_t bs = th / heads;                  // 0..4095
  const int s = (int)(bs & (S_ - 1));
  const size_t base = bs * (size_t)(heads * HD_) + (size_t)h * HD_;
  const float c  = cosT[s * HD_ + d];
  const float sn = sinT[s * HD_ + d];
  const float x0 = (float)X[base + d];
  const float x1 = (float)X[base + d + 64];
  X[base + d]      = (_Float16)((x0 * c - x1 * sn) * scale);
  X[base + d + 64] = (_Float16)((x1 * c + x0 * sn) * scale);
}

// ---------------------------------------------------------------------------
// Flash attention (non-causal, GQA). Block = (b, h, 128 q-rows); 8 waves,
// each wave owns one 16-row Q tile; loop over 64-key chunks in LDS.
// ---------------------------------------------------------------------------
__global__ __launch_bounds__(256)
void attn_kernel(const _Float16* __restrict__ Q, const _Float16* __restrict__ Kc,
                 const _Float16* __restrict__ Vc, _Float16* __restrict__ O) {
  constexpr int LDK = 136, LDV = 72, LDP = 72;
  __shared__ _Float16 lsK[64 * LDK];       // [key][d]
  __shared__ _Float16 lsV[128 * LDV];      // [d][key]  (transposed)
  __shared__ _Float16 lsP[8 * 16 * LDP];   // per-wave P scratch [m][key]

  const int tid = threadIdx.x, lane = tid & 31, wave = tid >> 5;
  const int nL = lane & 15, h2 = lane >> 4;
  const int b = blockIdx.z, h = blockIdx.y, g = h >> 2;
  const int q0 = blockIdx.x * 128 + wave * 16;
  _Float16* myP = &lsP[wave * 16 * LDP];

  // ---- Q fragments (A layout), 16 rows x 128 dims = 4 frags ----
  v16h qf[4];
  {
    const _Float16* qrow = Q + (size_t)(b * S_ + q0 + nL) * (H_ * HD_) + h * HD_;
#pragma unroll
    for (int kf = 0; kf < 4; ++kf) {
      union { v16h v; unsigned u[8]; } r;
#pragma unroll
      for (int j = 0; j < 8; ++j)
        r.u[j] = *(const unsigned*)(qrow + kf * 32 + KA(j, h2));
      qf[kf] = r.v;
    }
  }

  float mrow[8], lrow[8];
  v8f oacc[8];
#pragma unroll
  for (int i = 0; i < 8; ++i) { mrow[i] = -1e30f; lrow[i] = 0.f; }
#pragma unroll
  for (int t = 0; t < 8; ++t) oacc[t] = (v8f){0, 0, 0, 0, 0, 0, 0, 0};

  for (int kc = 0; kc < S_; kc += 64) {
    __syncthreads();
    // ---- stage K chunk [64 x 128] ----
#pragma unroll
    for (int i = 0; i < 4; ++i) {
      const int e = tid + i * 256;
      const int key = e >> 4, seg = e & 15;
      const uint4 v = *(const uint4*)(Kc +
          (size_t)(b * S_ + kc + key) * (KV_ * HD_) + g * HD_ + seg * 8);
      *(uint4*)&lsK[key * LDK + seg * 8] = v;
    }
    // ---- stage V chunk transposed [128 x 64] ----
#pragma unroll
    for (int i = 0; i < 4; ++i) {
      const int e = tid + i * 256;
      const int key = e >> 4, seg = e & 15;
      const uint4 v = *(const uint4*)(Vc +
          (size_t)(b * S_ + kc + key) * (KV_ * HD_) + g * HD_ + seg * 8);
      const _Float16* hv = (const _Float16*)&v;
#pragma unroll
      for (int u = 0; u < 8; ++u)
        lsV[(seg * 8 + u) * LDV + key] = hv[u];
    }
    __syncthreads();

    // ---- scores S = Q @ K^T  (16 x 64) ----
    v8f sc[4];
#pragma unroll
    for (int nt = 0; nt < 4; ++nt) {
      sc[nt] = (v8f){0, 0, 0, 0, 0, 0, 0, 0};
#pragma unroll
      for (int kf = 0; kf < 4; ++kf) {
        union { v16h v; unsigned u[8]; } r;
#pragma unroll
        for (int j = 0; j < 8; ++j)
          r.u[j] = *(const unsigned*)(&lsK[(nt * 16 + nL) * LDK + kf * 32 + KB(j, h2)]);
        sc[nt] = wmma16(qf[kf], r.v, sc[nt]);
      }
    }

    // ---- online softmax (row m = h2*8 + i spans 16 lanes of this half) ----
#pragma unroll
    for (int i = 0; i < 8; ++i) {
      float mx = fmaxf(fmaxf(sc[0][i], sc[1][i]), fmaxf(sc[2][i], sc[3][i]));
      mx = fmaxf(mx, __shfl_xor(mx, 1, 32));
      mx = fmaxf(mx, __shfl_xor(mx, 2, 32));
      mx = fmaxf(mx, __shfl_xor(mx, 4, 32));
      mx = fmaxf(mx, __shfl_xor(mx, 8, 32));
      const float mn = fmaxf(mrow[i], mx);
      const float corr = __expf(mrow[i] - mn);
      mrow[i] = mn;
      float rs = 0.f;
#pragma unroll
      for (int nt = 0; nt < 4; ++nt) {
        const float p = __expf(sc[nt][i] - mn);
        rs += p;
        myP[(h2 * 8 + i) * LDP + nt * 16 + nL] = (_Float16)p;
      }
      rs += __shfl_xor(rs, 1, 32);
      rs += __shfl_xor(rs, 2, 32);
      rs += __shfl_xor(rs, 4, 32);
      rs += __shfl_xor(rs, 8, 32);
      lrow[i] = lrow[i] * corr + rs;
#pragma unroll
      for (int t = 0; t < 8; ++t) oacc[t][i] *= corr;
    }
    __syncthreads();   // make P visible (C-layout -> A-layout round trip)

    // ---- O += P(16x64) @ V(64x128) ----
    v16h pf[2];
#pragma unroll
    for (int kf = 0; kf < 2; ++kf) {
      union { v16h v; unsigned u[8]; } r;
#pragma unroll
      for (int j = 0; j < 8; ++j)
        r.u[j] = *(const unsigned*)(myP + nL * LDP + kf * 32 + KA(j, h2));
      pf[kf] = r.v;
    }
#pragma unroll
    for (int dt = 0; dt < 8; ++dt) {
#pragma unroll
      for (int kf = 0; kf < 2; ++kf) {
        union { v16h v; unsigned u[8]; } r;
#pragma unroll
        for (int j = 0; j < 8; ++j)
          r.u[j] = *(const unsigned*)(&lsV[(dt * 16 + nL) * LDV + kf * 32 + KB(j, h2)]);
        oacc[dt] = wmma16(pf[kf], r.v, oacc[dt]);
      }
    }
  }

  // ---- epilogue: O = acc / l, f16 out at [b, s, h, d] ----
#pragma unroll
  for (int dt = 0; dt < 8; ++dt)
#pragma unroll
    for (int i = 0; i < 8; ++i) {
      const int row = q0 + h2 * 8 + i;
      const int col = h * HD_ + dt * 16 + nL;
      O[(size_t)(b * S_ + row) * (H_ * HD_) + col] =
          (_Float16)(oacc[dt][i] / lrow[i]);
    }
}

// ---------------------------------------------------------------------------
extern "C" void kernel_launch(void* const* d_in, const int* in_sizes, int n_in,
                              void* d_out, int out_size, void* d_ws, size_t ws_size,
                              hipStream_t stream) {
  (void)in_sizes; (void)n_in; (void)out_size; (void)ws_size;
  const float* hidden = (const float*)d_in[0];
  const float* cosT   = (const float*)d_in[1];
  const float* sinT   = (const float*)d_in[2];
  const float* Wq     = (const float*)d_in[3];
  const float* Wk     = (const float*)d_in[4];
  const float* Wv     = (const float*)d_in[5];
  const float* Wo     = (const float*)d_in[6];
  float* out = (float*)d_out;

  char* ws = (char*)d_ws;
  _Float16* Qh = (_Float16*)(ws);                                    // 16 MB
  _Float16* Kh = (_Float16*)(ws + (size_t)16 * 1024 * 1024);         //  4 MB
  _Float16* Vh = (_Float16*)(ws + (size_t)20 * 1024 * 1024);         //  4 MB
  _Float16* Ah = (_Float16*)(ws + (size_t)24 * 1024 * 1024);         // 16 MB

  const dim3 blk(256);

  // QKV projections (f32 in, f16 out)
  gemm_wmma_kernel<false, true><<<dim3(2048 / 128, TOK_ / 128), blk, 0, stream>>>(
      hidden, Wq, Qh, TOK_, 2048, 2048);
  gemm_wmma_kernel<false, true><<<dim3(512 / 128, TOK_ / 128), blk, 0, stream>>>(
      hidden, Wk, Kh, TOK_, 512, 2048);
  gemm_wmma_kernel<false, true><<<dim3(512 / 128, TOK_ / 128), blk, 0, stream>>>(
      hidden, Wv, Vh, TOK_, 512, 2048);

  // RoPE (fold 1/sqrt(HD) attention scale into Q)
  rope_kernel<<<(TOK_ * H_ * 64) / 256, blk, 0, stream>>>(
      Qh, cosT, sinT, H_, 0.08838834764831845f);
  rope_kernel<<<(TOK_ * KV_ * 64) / 256, blk, 0, stream>>>(
      Kh, cosT, sinT, KV_, 1.0f);

  // Flash attention
  attn_kernel<<<dim3(S_ / 128, H_, B_), blk, 0, stream>>>(Qh, Kh, Vh, Ah);

  // Output projection (f16 in, f32 out)
  gemm_wmma_kernel<true, false><<<dim3(2048 / 128, TOK_ / 128), blk, 0, stream>>>(
      Ah, Wo, out, TOK_, 2048, 2048);
}
